// MomNet_66795331387795
// MI455X (gfx1250) — matrix operand order repaired and checked by reading, not compiled, not run on hardware
//
#include <hip/hip_runtime.h>

// ---------------------------------------------------------------------------
// MomNet GNN on MI455X (gfx1250): bf16 storage, f32 accumulate, WMMA GEMMs.
// ---------------------------------------------------------------------------

typedef __attribute__((ext_vector_type(16))) __bf16 v16bf;
typedef __attribute__((ext_vector_type(8)))  __bf16 v8bf;
typedef __attribute__((ext_vector_type(4)))  __bf16 v4bf;
typedef __attribute__((ext_vector_type(8)))  float  v8f;

union BF16x16 { v16bf v; uint4 u[2]; };
union U4S8    { uint4 u; unsigned short s[8]; };

#define N_NODES 50000
#define N_EDGES 400000
#define E2      800000
#define HID     128

#define BM 64
#define BN 128
#define BK 32

// ---------------------------------------------------------------------------
// WMMA-tiled GEMM:  C[M x N] = act(A[M x K] @ W[K x N] + bias) (+ resid)
//   GATHER=1: K==256 and logical row r of A is
//             concat( feat[idxA[r]][0:128], feat[idxB[r]][0:128] ), feat
//             stride 128 (gather-concat fused into the LDS A-tile load).
//   N is a multiple of 16 and == BN per launch here (N=128, grid.x=1).
// Block: 256 threads = 8 waves laid out 4(M) x 2(N); each wave computes a
// 16x64 strip: one A fragment, four B fragments, 4 WMMAs per 32-deep K-step.
// B is staged TRANSPOSED in LDS so every lane's fragment is two runs of 16 contiguous
// bytes (2x ds_load_b128); A fragments are 2x16B contiguous per lane.
// ---------------------------------------------------------------------------
template <int GATHER>
__global__ __launch_bounds__(256)
void k_gemm(const __bf16* __restrict__ A,
            const int*    __restrict__ idxA,
            const int*    __restrict__ idxB,
            const __bf16* __restrict__ W,
            const float*  __restrict__ bias,
            const __bf16* __restrict__ resid,
            __bf16*       __restrict__ out16,
            float*        __restrict__ out32,
            int M, int N, int K, int relu)
{
    __shared__ __attribute__((aligned(16))) __bf16 As[BM * BK];      // [m][k], stride 32
    __shared__ __attribute__((aligned(16))) __bf16 BsT[BN * BK];     // [n][k], stride 32

    const int bm   = blockIdx.y * BM;
    const int bn   = blockIdx.x * BN;
    const int tid  = threadIdx.x;
    const int wave = tid >> 5;
    const int lane = tid & 31;
    const int tm   = (wave & 3) * 16;    // wave M offset in block tile
    const int tn   = (wave >> 2) * 64;   // wave N offset in block tile

    v8f acc0 = {}, acc1 = {}, acc2 = {}, acc3 = {};

    // cooperative-load thread mapping (computed once)
    const int ar = tid >> 2;             // A: row 0..63
    const int ac = (tid & 3) * 8;        // A: k-chunk of 8
    const int bkp = tid >> 4;            // B: k-pair 0..15  (k = 2*bkp, 2*bkp+1)
    const int bc  = (tid & 15) * 8;      // B: n-chunk of 8

    const int arow  = bm + ar;
    const int arowc = (arow < M) ? arow : (M - 1);

    for (int k0 = 0; k0 < K; k0 += BK) {
        // ---- A tile: each thread loads 8 bf16 (16B) of one row ----
        {
            const int gk = k0 + ac;
            const __bf16* rp;
            if (GATHER) {
                const int node = (gk < 128) ? idxA[arowc] : idxB[arowc];
                rp = A + (long long)node * 128 + (gk & 127);
            } else {
                rp = A + (long long)arowc * K + gk;
            }
            uint4 v = *(const uint4*)rp;
            if (arow >= M) v = make_uint4(0, 0, 0, 0);
            *(uint4*)&As[ar * 32 + ac] = v;
        }
        // ---- B tile (transposed): thread loads 8 n's from 2 k-rows, ----
        // ---- packs (k,k+1) pairs, stores 8x b32 into BsT[n][k]       ----
        {
            const long long gb = (long long)(k0 + 2 * bkp) * N + (bn + bc);
            U4S8 w0, w1;
            w0.u = *(const uint4*)(W + gb);
            w1.u = *(const uint4*)(W + gb + N);
            #pragma unroll
            for (int j = 0; j < 8; ++j) {
                const unsigned pk = (unsigned)w0.s[j] | ((unsigned)w1.s[j] << 16);
                *(unsigned*)&BsT[(bc + j) * 32 + 2 * bkp] = pk;
            }
        }
        if (wave == 0 && k0 + BK < K)   // warm L2 for next K-step of W
            __builtin_prefetch(W + (long long)(k0 + BK) * N + bn, 0, 1);
        __syncthreads();

        // ---- fragments (documented CDNA5 16-bit layouts) ----
        // A 16x32: lanes 0-15 hold K 0-7 & 16-23, lanes 16-31 hold 8-15 & 24-31
        BF16x16 af;
        {
            const int mrow = lane & 15;
            const int ak   = (lane < 16) ? 0 : 8;
            const __bf16* ap = &As[(tm + mrow) * 32];
            af.u[0] = *(const uint4*)(ap + ak);
            af.u[1] = *(const uint4*)(ap + ak + 16);
        }
        // B 32x16: lanes 0-15 hold K 0-7 & 16-23, lanes 16-31 hold 8-15 & 24-31
        const int ncol = lane & 15;
        const int bk0  = (lane < 16) ? 0 : 8;
        #pragma unroll
        for (int nt = 0; nt < 4; ++nt) {
            BF16x16 bf;
            const __bf16* bp = &BsT[(tn + nt * 16 + ncol) * 32 + bk0];
            bf.u[0] = *(const uint4*)bp;
            bf.u[1] = *(const uint4*)(bp + 16);
            if (nt == 0) acc0 = __builtin_amdgcn_wmma_f32_16x16x32_bf16(false, af.v, false, bf.v, (short)0, acc0, false, false);
            if (nt == 1) acc1 = __builtin_amdgcn_wmma_f32_16x16x32_bf16(false, af.v, false, bf.v, (short)0, acc1, false, false);
            if (nt == 2) acc2 = __builtin_amdgcn_wmma_f32_16x16x32_bf16(false, af.v, false, bf.v, (short)0, acc2, false, false);
            if (nt == 3) acc3 = __builtin_amdgcn_wmma_f32_16x16x32_bf16(false, af.v, false, bf.v, (short)0, acc3, false, false);
        }
        __syncthreads();
    }

    // ---- epilogue: C/D layout -> VGPR i holds M = i + 8*lane[4], N = lane&15
    const int mbase = bm + tm + ((lane >> 4) * 8);
    #pragma unroll
    for (int nt = 0; nt < 4; ++nt) {
        const v8f acc = (nt == 0) ? acc0 : (nt == 1) ? acc1 : (nt == 2) ? acc2 : acc3;
        const int col = bn + tn + nt * 16 + (lane & 15);
        const float bv = bias ? bias[col] : 0.0f;
        #pragma unroll
        for (int i = 0; i < 8; ++i) {
            const int row = mbase + i;
            if (row >= M) continue;
            float v = acc[i] + bv;
            if (relu) v = fmaxf(v, 0.0f);
            const long long off = (long long)row * N + col;
            if (resid) v += (float)resid[off];
            if (out32) out32[off] = v;
            if (out16) out16[off] = (__bf16)v;
        }
    }
}

// ---------------------------------------------------------------------------
// Tiny-K first embedding layer (K = 6 or 4): relu(X @ W + b) -> bf16
// One thread per 4 output channels; W read as float4.
// ---------------------------------------------------------------------------
__global__ void k_emb(const float* __restrict__ X, int K, int mod,
                      const float* __restrict__ W, const float* __restrict__ b,
                      __bf16* __restrict__ out, long long M)
{
    const long long gid = (long long)blockIdx.x * 256 + threadIdx.x;
    if (gid >= M * 32) return;
    const long long r  = gid >> 5;
    const int       c4 = (int)(gid & 31) * 4;
    const long long rr = mod ? (r % mod) : r;
    const float4 bb = *(const float4*)(b + c4);
    float a0 = bb.x, a1 = bb.y, a2 = bb.z, a3 = bb.w;
    for (int k = 0; k < K; ++k) {
        const float xv = X[rr * K + k];
        const float4 w = *(const float4*)(W + k * 128 + c4);
        a0 += xv * w.x; a1 += xv * w.y; a2 += xv * w.z; a3 += xv * w.w;
    }
    v4bf o;
    o[0] = (__bf16)fmaxf(a0, 0.0f); o[1] = (__bf16)fmaxf(a1, 0.0f);
    o[2] = (__bf16)fmaxf(a2, 0.0f); o[3] = (__bf16)fmaxf(a3, 0.0f);
    *(v4bf*)(out + r * 128 + c4) = o;
}

__global__ void k_srcdst(const int* __restrict__ ei, int* __restrict__ src,
                         int* __restrict__ dst)
{
    const int e = blockIdx.x * 256 + threadIdx.x;
    if (e >= N_EDGES) return;
    const int s = ei[e], d = ei[N_EDGES + e];
    src[e] = s;            dst[e] = d;
    src[N_EDGES + e] = d;  dst[N_EDGES + e] = s;
}

__global__ void k_fill32(unsigned* __restrict__ p, unsigned v, long long n)
{
    const long long gid = (long long)blockIdx.x * 256 + threadIdx.x;
    if (gid < n) p[gid] = v;
}

__global__ void k_cvt_f2b(const float* __restrict__ in, __bf16* __restrict__ out,
                          long long n)   // n multiple of 4
{
    const long long gid = (long long)blockIdx.x * 256 + threadIdx.x;
    if (gid * 4 >= n) return;
    const float4 f = *(const float4*)(in + gid * 4);
    v4bf o; o[0] = (__bf16)f.x; o[1] = (__bf16)f.y; o[2] = (__bf16)f.z; o[3] = (__bf16)f.w;
    *(v4bf*)(out + gid * 4) = o;
}

// ordered-uint encoding for float atomic max
__device__ __forceinline__ unsigned fkey(float f) {
    unsigned u = __float_as_uint(f);
    return (f >= 0.0f) ? (u | 0x80000000u) : ~u;
}
__device__ __forceinline__ float funkey(unsigned u) {
    return (u & 0x80000000u) ? __uint_as_float(u & 0x7FFFFFFFu)
                             : __uint_as_float(~u);
}

// alpha[e,h] = dot(q[dst], k[src]+ee) / sqrt(32); segment max via atomicMax
__global__ void k_score(const __bf16* __restrict__ q, const __bf16* __restrict__ k,
                        const __bf16* __restrict__ ee, const int* __restrict__ src,
                        const int* __restrict__ dst, float* __restrict__ alpha,
                        unsigned* __restrict__ amax)
{
    const long long gid = (long long)blockIdx.x * 256 + threadIdx.x;
    if (gid >= (long long)E2 * 4) return;
    const long long e = gid >> 2;
    const int       h = (int)(gid & 3);
    const int s = src[e], d = dst[e];
    const v8bf* qv = (const v8bf*)(q  + (long long)d * 128 + h * 32);
    const v8bf* kv = (const v8bf*)(k  + (long long)s * 128 + h * 32);
    const v8bf* ev = (const v8bf*)(ee + e * 128 + h * 32);
    float a = 0.0f;
    #pragma unroll
    for (int ch = 0; ch < 4; ++ch) {
        const v8bf qq = qv[ch], kk = kv[ch], e8 = ev[ch];
        #pragma unroll
        for (int j = 0; j < 8; ++j)
            a += (float)qq[j] * ((float)kk[j] + (float)e8[j]);
    }
    a *= 0.17677669529663687f;   // 1/sqrt(32)
    alpha[gid] = a;
    atomicMax(&amax[(long long)d * 4 + h], fkey(a));
}

// ex = exp(alpha - amax[dst]); den[dst] += ex  (ex written in place of alpha)
__global__ void k_exp(float* __restrict__ alpha, const unsigned* __restrict__ amax,
                      const int* __restrict__ dst, float* __restrict__ den)
{
    const long long gid = (long long)blockIdx.x * 256 + threadIdx.x;
    if (gid >= (long long)E2 * 4) return;
    const long long e = gid >> 2;
    const int       h = (int)(gid & 3);
    const int d = dst[e];
    const float ex = __expf(alpha[gid] - funkey(amax[(long long)d * 4 + h]));
    alpha[gid] = ex;
    atomicAdd(&den[(long long)d * 4 + h], ex);
}

// outc[dst] += (v[src] + ee) * ex/(den[dst]+1e-16); outc pre-seeded with skip()
// One thread per (edge, 4-channel chunk): 8B bf16 loads, 4 atomic f32 adds.
__global__ void k_scatter(const float* __restrict__ ex, const float* __restrict__ den,
                          const __bf16* __restrict__ v, const __bf16* __restrict__ ee,
                          const int* __restrict__ src, const int* __restrict__ dst,
                          float* __restrict__ outc)
{
    const long long gid = (long long)blockIdx.x * 256 + threadIdx.x;
    if (gid >= (long long)E2 * 32) return;
    const long long e  = gid >> 5;
    const int       c4 = (int)(gid & 31) * 4;
    const int       h  = c4 >> 5;
    const int s = src[e], d = dst[e];
    const float w = ex[e * 4 + h] / (den[(long long)d * 4 + h] + 1e-16f);
    const v4bf vv = *(const v4bf*)(v + (long long)s * 128 + c4);
    const v4bf e4 = *(const v4bf*)(ee + e * 128 + c4);
    float* op = outc + (long long)d * 128 + c4;
    #pragma unroll
    for (int j = 0; j < 4; ++j)
        atomicAdd(op + j, ((float)vv[j] + (float)e4[j]) * w);
}

// LayerNorm(cn) -> cn16 ; nf32 += ln (residual, in place) ; nf16 mirror.
// One wave per 128-wide row, 4 values per lane, shuffle reductions (wave32).
__global__ __launch_bounds__(256)
void k_ln(const float* __restrict__ cn, const float* __restrict__ g,
          const float* __restrict__ b, __bf16* __restrict__ cn16,
          float* __restrict__ nf32, __bf16* __restrict__ nf16, int M)
{
    const int row  = blockIdx.x * 8 + (threadIdx.x >> 5);
    const int lane = threadIdx.x & 31;
    if (row >= M) return;
    const long long base = (long long)row * 128 + lane * 4;
    const float4 vv = *(const float4*)(cn + base);
    float v[4] = {vv.x, vv.y, vv.z, vv.w};
    float s = v[0] + v[1] + v[2] + v[3];
    #pragma unroll
    for (int off = 16; off; off >>= 1) s += __shfl_xor(s, off, 32);
    const float mean = s * (1.0f / 128.0f);
    float var = 0.0f;
    #pragma unroll
    for (int i = 0; i < 4; ++i) { const float d = v[i] - mean; var += d * d; }
    #pragma unroll
    for (int off = 16; off; off >>= 1) var += __shfl_xor(var, off, 32);
    const float rs = rsqrtf(var * (1.0f / 128.0f) + 1e-5f);
    const float4 gg = *(const float4*)(g + lane * 4);
    const float4 bb = *(const float4*)(b + lane * 4);
    const float4 no = *(const float4*)(nf32 + base);
    const float gA[4] = {gg.x, gg.y, gg.z, gg.w};
    const float bA[4] = {bb.x, bb.y, bb.z, bb.w};
    const float nA[4] = {no.x, no.y, no.z, no.w};
    float4 nnew; v4bf c16, n16;
    float* nw = &nnew.x;
    #pragma unroll
    for (int i = 0; i < 4; ++i) {
        const float ln = (v[i] - mean) * rs * gA[i] + bA[i];
        c16[i] = (__bf16)ln;
        nw[i]  = ln + nA[i];
        n16[i] = (__bf16)nw[i];
    }
    *(v4bf*)(cn16 + base)   = c16;
    *(float4*)(nf32 + base) = nnew;
    *(v4bf*)(nf16 + base)   = n16;
}

// Final N=1 head: out[r] = dot(h[r, 0:128], w) + b0  (wave per row)
__global__ __launch_bounds__(256)
void k_dot(const __bf16* __restrict__ h, const float* __restrict__ w,
           const float* __restrict__ b, float* __restrict__ out, int M)
{
    const int row  = blockIdx.x * 8 + (threadIdx.x >> 5);
    const int lane = threadIdx.x & 31;
    if (row >= M) return;
    const v4bf hv = *(const v4bf*)(h + (long long)row * 128 + lane * 4);
    const float4 wv = *(const float4*)(w + lane * 4);
    float s = (float)hv[0] * wv.x + (float)hv[1] * wv.y +
              (float)hv[2] * wv.z + (float)hv[3] * wv.w;
    #pragma unroll
    for (int off = 16; off; off >>= 1) s += __shfl_xor(s, off, 32);
    if (lane == 0) out[row] = s + b[0];
}

// ---------------------------------------------------------------------------
// Host orchestration
// ---------------------------------------------------------------------------
extern "C" void kernel_launch(void* const* d_in, const int* in_sizes, int n_in,
                              void* d_out, int out_size, void* d_ws, size_t ws_size,
                              hipStream_t stream)
{
    (void)in_sizes; (void)n_in; (void)out_size; (void)ws_size;

    // Input index map (setup_inputs order, dicts in source order, lin = {w,b}):
    //  0 x  1 edge_attr  2 edge_index
    //  3..6   node_emb (w0,b0,w1,b1)      7..10 edge_emb
    //  11..40 mom_tc[0..2] (q.w q.b k.w k.b v.w v.b e.w e.b sk.w sk.b) x10
    //  41..52 edge_mlp[0..2] (w0,b0,w1,b1) x4
    //  53..58 mom_ln (g,b) x3
    //  59..62 mom_reg (w0,b0,w1,b1)
    //  63..72 edge_tc      73..76 edge_cls
    const float* x_in  = (const float*)d_in[0];
    const float* ea_in = (const float*)d_in[1];
    const int*   ei    = (const int*)  d_in[2];
    auto F = [&](int i) { return (const float*)d_in[i]; };

    char* wp = (char*)d_ws;
    auto alloc = [&](size_t bytes) -> void* {
        void* r = (void*)wp;
        wp += (bytes + 255) & ~(size_t)255;
        return r;
    };

    int*      src   = (int*)     alloc((size_t)E2 * 4);
    int*      dst   = (int*)     alloc((size_t)E2 * 4);
    float*    nf32  = (float*)   alloc((size_t)N_NODES * HID * 4);
    __bf16*   nf16  = (__bf16*)  alloc((size_t)N_NODES * HID * 2);
    float*    cn32  = (float*)   alloc((size_t)N_NODES * HID * 4);
    __bf16*   cn16  = (__bf16*)  alloc((size_t)N_NODES * HID * 2);
    __bf16*   q16   = (__bf16*)  alloc((size_t)N_NODES * HID * 2);
    __bf16*   k16   = (__bf16*)  alloc((size_t)N_NODES * HID * 2);
    __bf16*   v16   = (__bf16*)  alloc((size_t)N_NODES * HID * 2);
    unsigned* amax  = (unsigned*)alloc((size_t)N_NODES * 4 * 4);
    float*    den   = (float*)   alloc((size_t)N_NODES * 4 * 4);
    float*    alpha = (float*)   alloc((size_t)E2 * 4 * 4);
    __bf16*   ee16  = (__bf16*)  alloc((size_t)E2 * HID * 2);
    __bf16*   efA   = (__bf16*)  alloc((size_t)E2 * HID * 2);
    __bf16*   efB   = (__bf16*)  alloc((size_t)E2 * HID * 2);
    __bf16*   h116  = (__bf16*)  alloc((size_t)E2 * HID * 2);

    auto cvt = [&](const float* f, __bf16* b, long long n) {
        k_cvt_f2b<<<(unsigned)((n / 4 + 255) / 256), 256, 0, stream>>>(f, b, n);
    };
    auto W16 = [&](int idx, int n) -> __bf16* {
        __bf16* w = (__bf16*)alloc((size_t)n * 2);
        cvt(F(idx), w, n);
        return w;
    };

    // weights -> bf16
    __bf16* w_ne2 = W16(5, 128 * 128);
    __bf16* w_ee2 = W16(9, 128 * 128);
    __bf16 *tc_q[4], *tc_k[4], *tc_v[4], *tc_e[4], *tc_s[4];
    const int tc_base[4] = {11, 21, 31, 63};            // mom_tc x3 + edge_tc
    for (int t = 0; t < 4; ++t) {
        const int b0 = tc_base[t];
        tc_q[t] = W16(b0 + 0, 128 * 128);
        tc_k[t] = W16(b0 + 2, 128 * 128);
        tc_v[t] = W16(b0 + 4, 128 * 128);
        tc_e[t] = W16(b0 + 6, 128 * 128);
        tc_s[t] = W16(b0 + 8, 128 * 128);
    }
    __bf16 *em_w1[3], *em_w2[3];
    for (int t = 0; t < 3; ++t) {
        em_w1[t] = W16(41 + t * 4, 256 * 128);
        em_w2[t] = W16(43 + t * 4, 128 * 128);
    }
    __bf16* w_mr1 = W16(59, 256 * 128);
    __bf16* w_ec1 = W16(73, 256 * 128);

    k_srcdst<<<(N_EDGES + 255) / 256, 256, 0, stream>>>(ei, src, dst);

    auto gemm = [&](const __bf16* A, const int* iA, const int* iB,
                    const __bf16* W, const float* bias, const __bf16* resid,
                    __bf16* o16, float* o32, int M, int K, int relu) {
        dim3 g((HID + BN - 1) / BN, (M + BM - 1) / BM);
        if (iA)
            k_gemm<1><<<g, 256, 0, stream>>>(A, iA, iB, W, bias, resid, o16, o32, M, HID, K, relu);
        else
            k_gemm<0><<<g, 256, 0, stream>>>(A, nullptr, nullptr, W, bias, resid, o16, o32, M, HID, K, relu);
    };

    // ---- node/edge embeddings ----
    {
        long long n = (long long)N_NODES * 32;
        k_emb<<<(unsigned)((n + 255) / 256), 256, 0, stream>>>(
            x_in, 6, 0, F(3), F(4), h116, N_NODES);
        gemm(h116, nullptr, nullptr, w_ne2, F(6), nullptr, nf16, nf32, N_NODES, 128, 1);
        n = (long long)E2 * 32;
        k_emb<<<(unsigned)((n + 255) / 256), 256, 0, stream>>>(
            ea_in, 4, N_EDGES, F(7), F(8), h116, E2);
        gemm(h116, nullptr, nullptr, w_ee2, F(10), nullptr, efA, nullptr, E2, 128, 1);
    }

    __bf16* ef  = efA;
    __bf16* efN = efB;

    auto tconv = [&](int t, const __bf16* xin, const __bf16* ein, float* out_f32) {
        const int b0 = tc_base[t];
        gemm(xin, nullptr, nullptr, tc_q[t], F(b0 + 1), nullptr, q16, nullptr, N_NODES, 128, 0);
        gemm(xin, nullptr, nullptr, tc_k[t], F(b0 + 3), nullptr, k16, nullptr, N_NODES, 128, 0);
        gemm(xin, nullptr, nullptr, tc_v[t], F(b0 + 5), nullptr, v16, nullptr, N_NODES, 128, 0);
        gemm(ein, nullptr, nullptr, tc_e[t], F(b0 + 7), nullptr, ee16, nullptr, E2, 128, 0);
        // seed output with the skip linear (f32 accumulator), scatter adds on top
        gemm(xin, nullptr, nullptr, tc_s[t], F(b0 + 9), nullptr, nullptr, out_f32, N_NODES, 128, 0);
        const long long ns = (long long)N_NODES * 4;
        k_fill32<<<(unsigned)((ns + 255) / 256), 256, 0, stream>>>(amax, 0u, ns);
        k_fill32<<<(unsigned)((ns + 255) / 256), 256, 0, stream>>>((unsigned*)den, 0u, ns);
        const long long na = (long long)E2 * 4;
        k_score<<<(unsigned)((na + 255) / 256), 256, 0, stream>>>(q16, k16, ee16, src, dst, alpha, amax);
        k_exp  <<<(unsigned)((na + 255) / 256), 256, 0, stream>>>(alpha, amax, dst, den);
        const long long nm = (long long)E2 * 32;
        k_scatter<<<(unsigned)((nm + 255) / 256), 256, 0, stream>>>(alpha, den, v16, ee16, src, dst, out_f32);
    };

    // ---- message-passing iterations ----
    for (int t = 0; t < 3; ++t) {
        tconv(t, nf16, ef, cn32);
        k_ln<<<(N_NODES + 7) / 8, 256, 0, stream>>>(cn32, F(53 + 2 * t), F(54 + 2 * t),
                                                    cn16, nf32, nf16, N_NODES);
        // edge MLP on concat(cn[src], cn[dst]) -> + residual ef
        gemm(cn16, src, dst, em_w1[t], F(42 + t * 4), nullptr, h116, nullptr, E2, 256, 1);
        gemm(h116, nullptr, nullptr, em_w2[t], F(44 + t * 4), ef, efN, nullptr, E2, 128, 1);
        __bf16* tmp = ef; ef = efN; efN = tmp;
    }

    // ---- momentum head: mlp(concat(nf[src0], nf[dst0])) ----
    float* out = (float*)d_out;
    gemm(nf16, ei, ei + N_EDGES, w_mr1, F(60), nullptr, h116, nullptr, N_EDGES, 256, 1);
    k_dot<<<(N_EDGES + 7) / 8, 256, 0, stream>>>(h116, F(61), F(62), out, N_EDGES);

    // ---- edge classifier: edge_tc tconv (no LN/residual) then head ----
    tconv(3, nf16, ef, cn32);
    cvt(cn32, cn16, (long long)N_NODES * HID);   // fn16
    gemm(cn16, ei, ei + N_EDGES, w_ec1, F(74), nullptr, h116, nullptr, N_EDGES, 256, 1);
    k_dot<<<(N_EDGES + 7) / 8, 256, 0, stream>>>(h116, F(75), F(76), out + N_EDGES, N_EDGES);
}
